// CrossAttention_4870492914069
// MI455X (gfx1250) — compile-verified, matrix operand order checked by reading, not run on hardware
//
#include <hip/hip_runtime.h>

#define HEADS_N 8
#define DHEAD   64
#define INNER_D 512
#define QDIM    1024
#define CDIM    768
#define ODIM    1024
#define BATCH   2
#define NQS     4096
#define NKS     4096
#define MROWS   (BATCH * NQS) /* 8192 */

typedef __attribute__((ext_vector_type(16))) __bf16        v16bf;
typedef __attribute__((ext_vector_type(8)))  float          v8f;
typedef __attribute__((ext_vector_type(8)))  unsigned short v8u;
typedef __attribute__((ext_vector_type(16))) unsigned short v16u;
typedef __attribute__((ext_vector_type(4)))  unsigned int   v4u;
typedef __attribute__((ext_vector_type(4)))  int            v4i;
typedef __attribute__((ext_vector_type(8)))  int            v8i;

#if defined(__has_builtin)
#if __has_builtin(__builtin_amdgcn_tensor_load_to_lds) && __has_builtin(__builtin_amdgcn_s_wait_tensorcnt)
#define USE_TDM 1
#endif
#endif
#ifndef USE_TDM
#define USE_TDM 0
#endif

// log2(e) folded into the 1/sqrt(64) attention scale (softmax done base-2)
#define SCALE2 0.18033688011112042f

static __device__ __forceinline__ unsigned short f2bf(float f) {
  __bf16 h = (__bf16)f;                          // hw v_cvt, RTNE
  return __builtin_bit_cast(unsigned short, h);
}

static __device__ __forceinline__ v8f wmma_bf16(v16u a, v16u b, v8f c) {
  return __builtin_amdgcn_wmma_f32_16x16x32_bf16(
      false, __builtin_bit_cast(v16bf, a),
      false, __builtin_bit_cast(v16bf, b),
      (short)0, c, false, false);
}

// A/B fragment loader per ISA 16-bit A 16x32 layout: lane (m = lane%16,
// half = lane/16), element i <-> k = 16*(i>=8) + 8*half + (i%8)
// => two contiguous 16-byte runs at col 8*half and 16+8*half.
static __device__ __forceinline__ v16u load_frag16(const unsigned short* rowptr, int half) {
  const unsigned short* p = rowptr + (half << 3);
  v8u lo = *(const v8u*)p;
  v8u hi = *(const v8u*)(p + 16);
  return __builtin_shufflevector(lo, hi, 0,1,2,3,4,5,6,7,8,9,10,11,12,13,14,15);
}

static __device__ __forceinline__ float4 f4max(float4 a, float4 b) {
  float4 r;
  r.x = fmaxf(a.x, b.x); r.y = fmaxf(a.y, b.y);
  r.z = fmaxf(a.z, b.z); r.w = fmaxf(a.w, b.w);
  return r;
}

#if USE_TDM
// One TDM 2-D tile load: d1 rows x d0 elements (bf16), global row stride
// `stride` elements, into LDS at byte offset lds_byte. D# per ISA §8.
static __device__ __forceinline__ void tdm_load_2d(const unsigned short* gsrc,
                                                   unsigned int lds_byte,
                                                   int d0, int d1, int stride) {
  unsigned long long ga = (unsigned long long)gsrc;
  v4u g0;
  g0.x = 1u;                                     // count=1, user descriptor
  g0.y = lds_byte;                               // lds_addr
  g0.z = (unsigned int)ga;                       // global_addr[31:0]
  g0.w = (unsigned int)(ga >> 32) | (2u << 30);  // global_addr[56:32] | type=2
  v8i g1;
  g1[0] = 0x00010000;          // workgroup_mask=0, data_size=1 (2 bytes)
  g1[1] = d0 << 16;            // tensor_dim0 low16 (bits 63:48)
  g1[2] = d1 << 16;            // tensor_dim1 low16 (bits 95:80)
  g1[3] = d0 << 16;            // tile_dim0 (bits 127:112)
  g1[4] = d1;                  // tile_dim1, tile_dim2=0
  g1[5] = stride;              // tensor_dim0_stride low32
  g1[6] = 0;
  g1[7] = 0;
  v4i z4 = {};
#if __clang_major__ >= 23
  v8i z8 = {};
  __builtin_amdgcn_tensor_load_to_lds(g0, g1, z4, z4, z8, 0);
#else
  __builtin_amdgcn_tensor_load_to_lds(g0, g1, z4, z4, 0);
#endif
}
static __device__ __forceinline__ unsigned int lds_offset(const void* p) {
  return (unsigned int)(unsigned long long)p;    // generic LDS addr low 32 = LDS offset
}
#endif

// ---------------- conversion kernels ----------------
__global__ void cvt_bf16(const float* __restrict__ in, unsigned short* __restrict__ out, int n) {
  int i = blockIdx.x * blockDim.x + threadIdx.x;
  if (i < n) out[i] = f2bf(in[i]);
}

// f32 [K][N] -> bf16 [N][K]
__global__ void cvtT_bf16(const float* __restrict__ in, unsigned short* __restrict__ out,
                          int K, int N) {
  int i = blockIdx.x * blockDim.x + threadIdx.x;
  if (i < K * N) {
    int k = i / N, n = i % N;
    out[(size_t)n * K + k] = f2bf(in[i]);
  }
}

// ---------------- register-tiled bf16 WMMA GEMM ----------------
// One wave computes a 64x32 C tile (4x2 grid of 16x16 WMMA tiles):
// per k-step, 6 fragment loads feed 8 WMMAs.
// MODE 0: bf16 C[M][N]   MODE 1: bf16 CT[N][M]   MODE 2: f32 C[M][N] + bias
template <int MODE>
__global__ void gemm_tiled(const unsigned short* __restrict__ A,
                           const unsigned short* __restrict__ BT,
                           void* __restrict__ Cv,
                           const float* __restrict__ bias,
                           int M, int N, int K) {
  int wave = (blockIdx.x * blockDim.x + threadIdx.x) >> 5;
  int lane = threadIdx.x & 31, lm = lane & 15, half = lane >> 4;
  int nt = N >> 5;
  int tm = wave / nt, tn = wave % nt;
  if (tm * 64 >= M) return;
  const unsigned short* arow = A  + (size_t)(tm * 64 + lm) * K;
  const unsigned short* brow = BT + (size_t)(tn * 32 + lm) * K;
  v8f c[4][2] = {};
  for (int k0 = 0; k0 < K; k0 += 32) {
    v16u a0 = load_frag16(arow + k0,            half);
    v16u a1 = load_frag16(arow + (size_t)16 * K + k0, half);
    v16u a2 = load_frag16(arow + (size_t)32 * K + k0, half);
    v16u a3 = load_frag16(arow + (size_t)48 * K + k0, half);
    v16u b0 = load_frag16(brow + k0,            half);
    v16u b1 = load_frag16(brow + (size_t)16 * K + k0, half);
    c[0][0] = wmma_bf16(a0, b0, c[0][0]);
    c[0][1] = wmma_bf16(a0, b1, c[0][1]);
    c[1][0] = wmma_bf16(a1, b0, c[1][0]);
    c[1][1] = wmma_bf16(a1, b1, c[1][1]);
    c[2][0] = wmma_bf16(a2, b0, c[2][0]);
    c[2][1] = wmma_bf16(a2, b1, c[2][1]);
    c[3][0] = wmma_bf16(a3, b0, c[3][0]);
    c[3][1] = wmma_bf16(a3, b1, c[3][1]);
  }
#pragma unroll
  for (int mi = 0; mi < 4; mi++) {
#pragma unroll
    for (int ni = 0; ni < 2; ni++) {
      union { v8f v; float f[8]; } r; r.v = c[mi][ni];
      int col = tn * 32 + ni * 16 + lm;
      int rowb = tm * 64 + mi * 16 + 8 * half;
      if (MODE == 0) {
        unsigned short* C = (unsigned short*)Cv;
#pragma unroll
        for (int i = 0; i < 8; i++) C[(size_t)(rowb + i) * N + col] = f2bf(r.f[i]);
      } else if (MODE == 1) {
        unsigned short* CT = (unsigned short*)Cv;
        union { v8u v; unsigned short s[8]; } pk;
#pragma unroll
        for (int i = 0; i < 8; i++) pk.s[i] = f2bf(r.f[i]);
        *(v8u*)(CT + (size_t)col * M + rowb) = pk.v;
      } else {
        float* C = (float*)Cv;
        float bv = bias[col];
#pragma unroll
        for (int i = 0; i < 8; i++) C[(size_t)(rowb + i) * N + col] = r.f[i] + bv;
      }
    }
  }
}

// ---------------- flash attention ----------------
// Q, K: bf16 [B*N][512] (head-interleaved).  Vt: bf16 [512][8192].
// A block owns one (b,h) and 8 consecutive 16-query tiles (one per wave).
// K/V 32-key tiles are DMA'd once per BLOCK by the Tensor Data Mover
// (wave 0 issues, double-buffered, 2 tiles in flight), shared by all 8
// waves via __syncthreads.  Softmax: logits staged in LDS as f32; each
// lane reads its full 32-value row (row max = pure VALU, no shuffles),
// exp2 in A-layout gives the pA fragment directly, row sums come from a
// ones-matrix WMMA, and the new maxes broadcast via a 16-float LDS line.
__global__ void attn16(const unsigned short* __restrict__ Q,
                       const unsigned short* __restrict__ Kb,
                       const unsigned short* __restrict__ Vt,
                       unsigned short* __restrict__ O) {
  __shared__ __align__(16) float pX[8][16 * 32];
  __shared__ __align__(16) float tLds[8][16];
#if USE_TDM
  __shared__ __align__(16) unsigned short kSh[2][32 * 64];
  __shared__ __align__(16) unsigned short vSh[2][64 * 32];
#endif
  int w    = threadIdx.x >> 5;
  int lane = threadIdx.x & 31, lm = lane & 15, half = lane >> 4;
  int qg = blockIdx.x & 31;              // 32 q-groups of 8 tiles per (b,h)
  int h  = (blockIdx.x >> 5) & 7;
  int b  = blockIdx.x >> 8;
  int qt = qg * 8 + w;

  const unsigned short* qrow = Q + (size_t)(b * NQS + qt * 16 + lm) * INNER_D + h * DHEAD;
  v16u a0 = load_frag16(qrow, half);
  v16u a1 = load_frag16(qrow + 32, half);

  union { v16u v; unsigned short s[16]; } onesU;
#pragma unroll
  for (int i = 0; i < 16; i++) onesU.s[i] = 0x3F80;   // bf16 1.0
  v16u onesB = onesU.v;

  v8f o0 = {}, o1 = {}, o2 = {}, o3 = {};
  float mC[8], l[8];
#pragma unroll
  for (int i = 0; i < 8; i++) { mC[i] = -1e30f; l[i] = 0.f; }
  float mA = -1e30f;

  const unsigned short* kbase = Kb + (size_t)b * NKS * INNER_D + h * DHEAD;
  const unsigned short* vbase = Vt + (size_t)h * DHEAD * MROWS + b * NKS;
#if USE_TDM
  unsigned int kofs[2], vofs[2];
  kofs[0] = lds_offset(&kSh[0][0]);  kofs[1] = lds_offset(&kSh[1][0]);
  vofs[0] = lds_offset(&vSh[0][0]);  vofs[1] = lds_offset(&vSh[1][0]);
  if (w == 0) {                                     // preload tiles 0 and 1
    tdm_load_2d(kbase, kofs[0], 64, 32, INNER_D);
    tdm_load_2d(vbase, vofs[0], 32, 64, MROWS);
    tdm_load_2d(kbase + (size_t)32 * INNER_D, kofs[1], 64, 32, INNER_D);
    tdm_load_2d(vbase + 32,                   vofs[1], 32, 64, MROWS);
  }
#endif

  for (int j0 = 0; j0 < NKS; j0 += 32) {
    v8f z = {};
    v8f s0, s1;
#if USE_TDM
    int buf = (j0 >> 5) & 1;
    if (w == 0) {
      if (j0 + 32 < NKS) __builtin_amdgcn_s_wait_tensorcnt(2);  // tile j0 landed
      else               __builtin_amdgcn_s_wait_tensorcnt(0);
    }
    __syncthreads();                                // tile j0 visible block-wide
    const unsigned short* kt = &kSh[buf][0];        // 32 rows(keys) x 64 cols(d)
    s0 = wmma_bf16(a1, load_frag16(kt + lm * 64 + 32, half),
                   wmma_bf16(a0, load_frag16(kt + lm * 64, half), z));
    s1 = wmma_bf16(a1, load_frag16(kt + (16 + lm) * 64 + 32, half),
                   wmma_bf16(a0, load_frag16(kt + (16 + lm) * 64, half), z));
#else
    const unsigned short* k0r = kbase + (size_t)(j0 + lm) * INNER_D;
    const unsigned short* k1r = k0r + 16 * INNER_D;
    if (j0 + 32 < NKS) __builtin_prefetch(k0r + 32 * INNER_D, 0, 0);
    s0 = wmma_bf16(a1, load_frag16(k0r + 32, half),
                   wmma_bf16(a0, load_frag16(k0r, half), z));
    s1 = wmma_bf16(a1, load_frag16(k1r + 32, half),
                   wmma_bf16(a0, load_frag16(k1r, half), z));
#endif
    // --- stage scaled logits (C layout) in LDS as f32 ---
    union { v8f v; float f[8]; } u0, u1;
    u0.v = s0; u1.v = s1;
#pragma unroll
    for (int i = 0; i < 8; i++) {
      int row = i + 8 * half;
      pX[w][row * 32 + lm]      = u0.f[i] * SCALE2;
      pX[w][row * 32 + 16 + lm] = u1.f[i] * SCALE2;
    }
    asm volatile("s_wait_dscnt 0" ::: "memory");
    // --- each lane reads its FULL row: max needs no cross-lane op ---
    const float* prow = &pX[w][lm * 32];
    float4 q0 = *(const float4*)(prow +  0);
    float4 q1 = *(const float4*)(prow +  4);
    float4 q2 = *(const float4*)(prow +  8);
    float4 q3 = *(const float4*)(prow + 12);
    float4 q4 = *(const float4*)(prow + 16);
    float4 q5 = *(const float4*)(prow + 20);
    float4 q6 = *(const float4*)(prow + 24);
    float4 q7 = *(const float4*)(prow + 28);
    float4 mx = f4max(f4max(f4max(q0, q1), f4max(q2, q3)),
                      f4max(f4max(q4, q5), f4max(q6, q7)));
    float t = fmaxf(fmaxf(mx.x, mx.y), fmaxf(mx.z, mx.w));
    float nmA = fmaxf(mA, t);
    mA = nmA;
    tLds[w][lm] = nmA;                              // broadcast new max to C layout
    // --- own A-fragment halves from the loaded row, exp2 -> pA ---
    float4 s0a = half ? q2 : q0;                    // k = 8*half .. +7
    float4 s0b = half ? q3 : q1;
    float4 s1a = half ? q6 : q4;                    // k = 16+8*half .. +7
    float4 s1b = half ? q7 : q5;
    float own[16] = { s0a.x, s0a.y, s0a.z, s0a.w, s0b.x, s0b.y, s0b.z, s0b.w,
                      s1a.x, s1a.y, s1a.z, s1a.w, s1b.x, s1b.y, s1b.z, s1b.w };
    union { v16u v; unsigned short s[16]; } pu;
#pragma unroll
    for (int i = 0; i < 16; i++) pu.s[i] = f2bf(exp2f(own[i] - nmA));
    v16u pA = pu.v;
    asm volatile("s_wait_dscnt 0" ::: "memory");
    float4 nc0 = *(const float4*)&tLds[w][8 * half];
    float4 nc1 = *(const float4*)&tLds[w][8 * half + 4];
    float nmC[8] = { nc0.x, nc0.y, nc0.z, nc0.w, nc1.x, nc1.y, nc1.z, nc1.w };
    // --- row sums via ones-matrix WMMA (lands in C layout) ---
    v8f sums = wmma_bf16(pA, onesB, z);
    union { v8f v; float f[8]; } su; su.v = sums;
    {
      union { v8f v; float f[8]; } t0, t1, t2, t3;
      t0.v = o0; t1.v = o1; t2.v = o2; t3.v = o3;
#pragma unroll
      for (int i = 0; i < 8; i++) {
        float al = exp2f(mC[i] - nmC[i]);
        mC[i] = nmC[i];
        l[i] = l[i] * al + su.f[i];
        t0.f[i] *= al; t1.f[i] *= al; t2.f[i] *= al; t3.f[i] *= al;
      }
      o0 = t0.v; o1 = t1.v; o2 = t2.v; o3 = t3.v;
    }
    // --- P @ V ---
#if USE_TDM
    const unsigned short* vt = &vSh[buf][0];        // 64 rows(d) x 32 cols(keys)
    o0 = wmma_bf16(pA, load_frag16(vt + lm * 32,        half), o0);
    o1 = wmma_bf16(pA, load_frag16(vt + (16 + lm) * 32, half), o1);
    o2 = wmma_bf16(pA, load_frag16(vt + (32 + lm) * 32, half), o2);
    o3 = wmma_bf16(pA, load_frag16(vt + (48 + lm) * 32, half), o3);
    __syncthreads();                                // all waves done with buf
    if (w == 0 && j0 + 64 < NKS) {                  // refill buf with tile j0+64
      tdm_load_2d(kbase + (size_t)(j0 + 64) * INNER_D, kofs[buf], 64, 32, INNER_D);
      tdm_load_2d(vbase + (j0 + 64),                   vofs[buf], 32, 64, MROWS);
    }
#else
    const unsigned short* vb = vbase + (size_t)lm * MROWS + j0;
    o0 = wmma_bf16(pA, load_frag16(vb,              half), o0);
    o1 = wmma_bf16(pA, load_frag16(vb + 16 * MROWS, half), o1);
    o2 = wmma_bf16(pA, load_frag16(vb + 32 * MROWS, half), o2);
    o3 = wmma_bf16(pA, load_frag16(vb + 48 * MROWS, half), o3);
#endif
  }

  union { v8f v; float f[8]; } r0, r1, r2, r3;
  r0.v = o0; r1.v = o1; r2.v = o2; r3.v = o3;
  float inv[8];
#pragma unroll
  for (int i = 0; i < 8; i++) inv[i] = 1.0f / l[i];
  size_t rowbase = (size_t)(b * NQS + qt * 16);
#pragma unroll
  for (int i = 0; i < 8; i++) {
    size_t orow = (rowbase + i + 8 * half) * INNER_D + h * DHEAD + lm;
    O[orow]      = f2bf(r0.f[i] * inv[i]);
    O[orow + 16] = f2bf(r1.f[i] * inv[i]);
    O[orow + 32] = f2bf(r2.f[i] * inv[i]);
    O[orow + 48] = f2bf(r3.f[i] * inv[i]);
  }
}

// ---------------- host launcher ----------------
extern "C" void kernel_launch(void* const* d_in, const int* in_sizes, int n_in,
                              void* d_out, int out_size, void* d_ws, size_t ws_size,
                              hipStream_t stream) {
  const float* x     = (const float*)d_in[0];
  const float* ctx   = (const float*)d_in[1];
  const float* w_q   = (const float*)d_in[2];
  const float* w_k   = (const float*)d_in[3];
  const float* w_v   = (const float*)d_in[4];
  const float* w_out = (const float*)d_in[5];
  const float* b_out = (const float*)d_in[6];
  float* out = (float*)d_out;

  char* ws = (char*)d_ws;
  size_t off = 0;
  auto alloc = [&](size_t elems) {
    void* p = ws + off;
    off = (off + elems * 2 + 255) & ~(size_t)255;
    return (unsigned short*)p;
  };
  unsigned short* xb  = alloc((size_t)MROWS * QDIM);    // reused for Ob
  unsigned short* cb  = alloc((size_t)MROWS * CDIM);
  unsigned short* wqT = alloc((size_t)INNER_D * QDIM);
  unsigned short* wkT = alloc((size_t)INNER_D * CDIM);
  unsigned short* wvT = alloc((size_t)INNER_D * CDIM);
  unsigned short* woT = alloc((size_t)ODIM * INNER_D);
  unsigned short* Qb  = alloc((size_t)MROWS * INNER_D);
  unsigned short* Kp  = alloc((size_t)MROWS * INNER_D);
  unsigned short* Vt  = alloc((size_t)INNER_D * MROWS);
  unsigned short* Ob  = xb;

  const int T = 256;
  {
    int n = MROWS * QDIM;
    cvt_bf16<<<(n + T - 1) / T, T, 0, stream>>>(x, xb, n);
  }
  {
    int n = MROWS * CDIM;
    cvt_bf16<<<(n + T - 1) / T, T, 0, stream>>>(ctx, cb, n);
  }
  cvtT_bf16<<<(QDIM * INNER_D + T - 1) / T, T, 0, stream>>>(w_q, wqT, QDIM, INNER_D);
  cvtT_bf16<<<(CDIM * INNER_D + T - 1) / T, T, 0, stream>>>(w_k, wkT, CDIM, INNER_D);
  cvtT_bf16<<<(CDIM * INNER_D + T - 1) / T, T, 0, stream>>>(w_v, wvT, CDIM, INNER_D);
  cvtT_bf16<<<(INNER_D * ODIM + T - 1) / T, T, 0, stream>>>(w_out, woT, INNER_D, ODIM);

  auto gblocks = [&](int M, int N) { return ((M / 64) * (N / 32) * 32 + T - 1) / T; };
  gemm_tiled<0><<<gblocks(MROWS, INNER_D), T, 0, stream>>>(xb, wqT, (void*)Qb, nullptr,
                                                           MROWS, INNER_D, QDIM);
  gemm_tiled<0><<<gblocks(MROWS, INNER_D), T, 0, stream>>>(cb, wkT, (void*)Kp, nullptr,
                                                           MROWS, INNER_D, CDIM);
  gemm_tiled<1><<<gblocks(MROWS, INNER_D), T, 0, stream>>>(cb, wvT, (void*)Vt, nullptr,
                                                           MROWS, INNER_D, CDIM);

  attn16<<<512, T, 0, stream>>>(Qb, Kp, Vt, Ob);   // 512 blocks x 8 waves

  gemm_tiled<2><<<gblocks(MROWS, ODIM), T, 0, stream>>>(Ob, woT, (void*)out, b_out,
                                                        MROWS, ODIM, INNER_D);
}
